// SepConv2d_60009283059776
// MI455X (gfx1250) — matrix operand order repaired and checked by reading, not compiled, not run on hardware
//
#include <hip/hip_runtime.h>

// ---------------------------------------------------------------------------
// SNN 2-layer pipeline for MI455X (gfx1250, wave32, WMMA f32_16x16x32_f16).
// Exploits the pad-47 structural sparsity: conv2 computed only on its 18x18
// live patch; final output written as zeros + a 6x6 scattered pooled patch.
// ---------------------------------------------------------------------------

typedef __attribute__((ext_vector_type(16))) _Float16 v16h;
typedef __attribute__((ext_vector_type(8)))  float    v8f;
typedef __attribute__((ext_vector_type(4)))  float    f32x4;

#define T_STEPS 15
#define C_IN    64
#define C_OUT   256
#define THR1    10.0f
#define THR2    1.0f

// workspace element counts
#define XCL1_ELEMS (T_STEPS*36*36*64)     // padded x, channels-last f16
#define W1R_ELEMS  (25*256*64)            // w1 as [kh][kw][co][ci] f16
#define W2R_ELEMS  (4*256*256)            // w2 as [kh][kw][co][ci] f16
#define POT1_ELEMS (T_STEPS*256*1024)     // (t,co,32,32) f32
#define POOL1_ELEMS (T_STEPS*256*289)     // (t,co,17,17) f32 (binary)
#define XCL2_ELEMS (T_STEPS*20*20*256)    // padded pooled spikes, channels-last f16
#define POT2_ELEMS (T_STEPS*256*324)      // (t,co,18,18) f32

// ---------------------------------------------------------------------------
// Packing kernels
// ---------------------------------------------------------------------------
__global__ void pack_xcl1(const float* __restrict__ x, _Float16* __restrict__ xcl) {
    int i = blockIdx.x * blockDim.x + threadIdx.x;
    if (i >= XCL1_ELEMS) return;
    int ci = i & 63;
    int r  = i >> 6;
    int wp = r % 36; r /= 36;
    int hp = r % 36; int t = r / 36;
    int h = hp - 2, w = wp - 2;
    float v = 0.0f;
    if (h >= 0 && h < 32 && w >= 0 && w < 32)
        v = x[((t * 64 + ci) * 32 + h) * 32 + w];
    xcl[i] = (_Float16)v;
}

__global__ void pack_w1(const float* __restrict__ w1, _Float16* __restrict__ w1r) {
    int i = blockIdx.x * blockDim.x + threadIdx.x;
    if (i >= W1R_ELEMS) return;
    int ci = i & 63;
    int r  = i >> 6;
    int co = r & 255; r >>= 8;
    int kw = r % 5;   int kh = r / 5;
    w1r[i] = (_Float16)w1[((co * 64 + ci) * 5 + kh) * 5 + kw];
}

__global__ void pack_w2(const float* __restrict__ w2, _Float16* __restrict__ w2r) {
    int i = blockIdx.x * blockDim.x + threadIdx.x;
    if (i >= W2R_ELEMS) return;
    int ci = i & 255;
    int r  = i >> 8;
    int co = r & 255; r >>= 8;
    int kw = r & 1;   int kh = r >> 1;
    w2r[i] = (_Float16)w2[((co * 256 + ci) * 2 + kh) * 2 + kw];
}

__global__ void pack_xcl2(const float* __restrict__ pooled, _Float16* __restrict__ xcl2) {
    int i = blockIdx.x * blockDim.x + threadIdx.x;
    if (i >= XCL2_ELEMS) return;
    int c  = i & 255;
    int r  = i >> 8;
    int wp = r % 20; r /= 20;
    int hp = r % 20; int t = r / 20;
    int p = hp - 1, q = wp - 1;
    float v = 0.0f;
    if (p >= 0 && p < 17 && q >= 0 && q < 17)
        v = pooled[((t * 256 + c) * 289) + p * 17 + q];
    xcl2[i] = (_Float16)v;
}

// ---------------------------------------------------------------------------
// conv1: implicit GEMM, M=co(16) x N=16 spatial, K = (kh,kw) x 2 ci-chunks of 32.
// Fragment layouts per CDNA5 ISA 7.12.2 (wave32):
//   A 16x32 f16: lane<16 -> row M=lane, K in {0..7} u {16..23}; lane>=16 shifts K by 8.
//   B 32x16 f16: lane<16 -> col N=lane, K in {0..7} u {16..23} as A; lane>=16 shifts K by 8.
//   D 16x16 f32: VGPR r -> row r (lanes<16) / r+8 (lanes>=16), col = lane&15.
// Fire (thr=10) fused into the store.
// ---------------------------------------------------------------------------
__global__ __launch_bounds__(256) void conv1_wmma(const _Float16* __restrict__ xcl,
                                                  const _Float16* __restrict__ w1r,
                                                  float* __restrict__ pot1) {
    const int lane   = threadIdx.x & 31;
    const int wid    = threadIdx.x >> 5;
    const int spTile = blockIdx.x;                 // 0..63  (row h0, half-row of 16 cols)
    const int coTile = blockIdx.y * 8 + wid;       // 0..15
    const int t      = blockIdx.z;                 // 0..14
    const int h0     = spTile >> 1;
    const int w0     = (spTile & 1) * 16;
    const int nl     = lane & 15;
    const bool hi    = lane >= 16;
    const int co     = coTile * 16 + nl;           // A row for this lane
    const int wcol   = w0 + nl;                    // B column (spatial) for this lane
    const int klo8   = hi ? 8 : 0;
    const int klo16  = hi ? 16 : 0;

    v8f acc = {};
    for (int kh = 0; kh < 5; ++kh) {
        if (kh < 4)
            __builtin_prefetch(xcl + (((t * 36 + (h0 + kh + 1)) * 36) + wcol) * 64, 0, 1);
        for (int kw = 0; kw < 5; ++kw) {
            const _Float16* xp = xcl + (((t * 36 + (h0 + kh)) * 36) + (wcol + kw)) * 64;
            const _Float16* wp = w1r + (((kh * 5 + kw) * 256 + co) * 64);
#pragma unroll
            for (int cb = 0; cb < 64; cb += 32) {
                v16h a, b;
                f32x4* av = reinterpret_cast<f32x4*>(&a);
                av[0] = *reinterpret_cast<const f32x4*>(wp + cb + klo8);
                av[1] = *reinterpret_cast<const f32x4*>(wp + cb + klo8 + 16);
                f32x4* bv = reinterpret_cast<f32x4*>(&b);
                bv[0] = *reinterpret_cast<const f32x4*>(xp + cb + klo8);
                bv[1] = *reinterpret_cast<const f32x4*>(xp + cb + klo8 + 16);
                acc = __builtin_amdgcn_wmma_f32_16x16x32_f16(false, a, false, b,
                                                             (short)0, acc, false, false);
            }
        }
    }
    const int m0 = hi ? 8 : 0;
    float* outp = pot1 + ((size_t)(t * 256 + coTile * 16) * 1024) + h0 * 32 + w0 + nl;
#pragma unroll
    for (int r = 0; r < 8; ++r) {
        float v = acc[r];
        outp[(size_t)(m0 + r) * 1024] = (v > THR1) ? v : 0.0f;
    }
}

// ---------------------------------------------------------------------------
// conv2 on the live 18x18 patch. Input xcl2 is [t][20][20][256] (pad 1 + guard
// row), output pot2 is (t,256,324). Fire (thr=1) fused.
// ---------------------------------------------------------------------------
__global__ __launch_bounds__(256) void conv2_wmma(const _Float16* __restrict__ xcl2,
                                                  const _Float16* __restrict__ w2r,
                                                  float* __restrict__ pot2) {
    const int lane   = threadIdx.x & 31;
    const int wid    = threadIdx.x >> 5;
    const int nTile  = blockIdx.x;                 // 0..20 (21*16 = 336 >= 324)
    const int coTile = blockIdx.y * 8 + wid;       // 0..15
    const int t      = blockIdx.z;
    const int nl     = lane & 15;
    const bool hi    = lane >= 16;
    const int n      = nTile * 16 + nl;            // spatial index 0..335
    const int gy     = n / 18;                     // up to 18 (guard rows are zero)
    const int gx     = n % 18;
    const int co     = coTile * 16 + nl;
    const int klo8   = hi ? 8 : 0;
    const int klo16  = hi ? 16 : 0;

    v8f acc = {};
#pragma unroll
    for (int kh = 0; kh < 2; ++kh) {
#pragma unroll
        for (int kw = 0; kw < 2; ++kw) {
            const _Float16* xp = xcl2 + ((size_t)((t * 20 + (gy + kh)) * 20 + (gx + kw))) * 256;
            const _Float16* wp = w2r + ((size_t)((kh * 2 + kw) * 256 + co)) * 256;
            for (int cb = 0; cb < 256; cb += 32) {
                v16h a, b;
                f32x4* av = reinterpret_cast<f32x4*>(&a);
                av[0] = *reinterpret_cast<const f32x4*>(wp + cb + klo8);
                av[1] = *reinterpret_cast<const f32x4*>(wp + cb + klo8 + 16);
                f32x4* bv = reinterpret_cast<f32x4*>(&b);
                bv[0] = *reinterpret_cast<const f32x4*>(xp + cb + klo8);
                bv[1] = *reinterpret_cast<const f32x4*>(xp + cb + klo8 + 16);
                acc = __builtin_amdgcn_wmma_f32_16x16x32_f16(false, a, false, b,
                                                             (short)0, acc, false, false);
            }
        }
    }
    if (n < 324) {
        const int m0 = hi ? 8 : 0;
        float* outp = pot2 + ((size_t)(t * 256 + coTile * 16) * 324) + n;
#pragma unroll
        for (int r = 0; r < 8; ++r) {
            float v = acc[r];
            outp[(size_t)(m0 + r) * 324] = (v > THR2) ? v : 0.0f;
        }
    }
}

// ---------------------------------------------------------------------------
// Pointwise inhibition, in place. One 256-thread block per spatial position.
// pot layout: [T][256][HW]. Matches jnp semantics: per (h,w), winner channel =
// argmax (first occurrence) at t_first = clip(T - #spiking-timesteps, 0, T-1);
// keep only winner channel, gated by spike-at-last-timestep.
// ---------------------------------------------------------------------------
__global__ __launch_bounds__(256) void inhibit_inplace(float* __restrict__ pot, int HW) {
    const int pos = blockIdx.x;
    const int c   = threadIdx.x;
    __shared__ float s_v[256];
    __shared__ int   s_i[256];
    __shared__ float s_mx[16];
    __shared__ int   s_am[16];
    __shared__ int   s_winner;
    __shared__ float s_spiked;

    for (int t = 0; t < T_STEPS; ++t) {
        s_v[c] = pot[((size_t)(t * 256 + c)) * HW + pos];
        s_i[c] = c;
        __syncthreads();
        for (int off = 128; off > 0; off >>= 1) {
            if (c < off) {
                float v2 = s_v[c + off]; int i2 = s_i[c + off];
                if (v2 > s_v[c] || (v2 == s_v[c] && i2 < s_i[c])) { s_v[c] = v2; s_i[c] = i2; }
            }
            __syncthreads();
        }
        if (c == 0) { s_mx[t] = s_v[0]; s_am[t] = s_i[0]; }
        __syncthreads();
    }
    if (c == 0) {
        int cnt = 0;
        for (int t = 0; t < T_STEPS; ++t) cnt += (s_mx[t] > 0.0f) ? 1 : 0;
        int tf = T_STEPS - cnt;
        if (tf < 0) tf = 0;
        if (tf > T_STEPS - 1) tf = T_STEPS - 1;
        s_winner = s_am[tf];
        s_spiked = (s_mx[T_STEPS - 1] > 0.0f) ? 1.0f : 0.0f;
    }
    __syncthreads();
    const float coef = (c == s_winner) ? s_spiked : 0.0f;
    for (int t = 0; t < T_STEPS; ++t) {
        size_t idx = ((size_t)(t * 256 + c)) * HW + pos;
        pot[idx] = pot[idx] * coef;
    }
}

// ---------------------------------------------------------------------------
// binarize(pot1) + 2x2/2 maxpool (pad 1): (t,256,32,32) -> (t,256,17,17)
// ---------------------------------------------------------------------------
__global__ void maxpool1(const float* __restrict__ pot1, float* __restrict__ pooled) {
    int i = blockIdx.x * blockDim.x + threadIdx.x;
    if (i >= T_STEPS * 256 * 289) return;
    int jx = i % 17;
    int jy = (i / 17) % 17;
    int c  = (i / 289) % 256;
    int t  = i / (289 * 256);
    const float* base = pot1 + ((size_t)(t * 256 + c)) * 1024;
    float m = 0.0f;
    for (int dy = 0; dy < 2; ++dy) {
        int yy = 2 * jy - 1 + dy;
        if (yy < 0 || yy >= 32) continue;
        for (int dx = 0; dx < 2; ++dx) {
            int xx = 2 * jx - 1 + dx;
            if (xx < 0 || xx >= 32) continue;
            float s = (base[yy * 32 + xx] > 0.0f) ? 1.0f : 0.0f;
            m = fmaxf(m, s);
        }
    }
    pooled[i] = m;
}

__global__ void zero_f32(float* __restrict__ p, int n) {
    int i = blockIdx.x * blockDim.x + threadIdx.x;
    if (i < n) p[i] = 0.0f;
}

// ---------------------------------------------------------------------------
// binarize(pot2) + 4x4/4 maxpool (pad 1) + pad 2, scattered into the 6x6 live
// region (rows/cols 13..18) of the zero-filled (t,256,32,32) output.
// ---------------------------------------------------------------------------
__global__ void final_pool(const float* __restrict__ pot2, float* __restrict__ out) {
    int i = blockIdx.x * blockDim.x + threadIdx.x;
    if (i >= T_STEPS * 256 * 36) return;
    int jx = (i % 6) + 11;
    int jy = ((i / 6) % 6) + 11;
    int c  = (i / 36) % 256;
    int t  = i / (36 * 256);
    const float* base = pot2 + ((size_t)(t * 256 + c)) * 324;
    float m = 0.0f;
    for (int dy = 0; dy < 4; ++dy) {
        int ly = jy * 4 - 47 + dy;          // local conv2 coord (grid offset 46, pad 1)
        if (ly < 0 || ly >= 18) continue;
        for (int dx = 0; dx < 4; ++dx) {
            int lx = jx * 4 - 47 + dx;
            if (lx < 0 || lx >= 18) continue;
            float s = (base[ly * 18 + lx] > 0.0f) ? 1.0f : 0.0f;
            m = fmaxf(m, s);
        }
    }
    out[((size_t)(t * 256 + c)) * 1024 + (2 + jy) * 32 + (2 + jx)] = m;
}

// ---------------------------------------------------------------------------
extern "C" void kernel_launch(void* const* d_in, const int* in_sizes, int n_in,
                              void* d_out, int out_size, void* d_ws, size_t ws_size,
                              hipStream_t stream) {
    (void)in_sizes; (void)n_in; (void)out_size; (void)ws_size;
    const float* x  = (const float*)d_in[0];
    const float* w1 = (const float*)d_in[1];
    const float* w2 = (const float*)d_in[2];
    float* out = (float*)d_out;

    char* ws = (char*)d_ws;
    size_t off = 0;
    auto carve = [&](size_t bytes) -> void* {
        void* p = ws + off;
        off = (off + bytes + 255) & ~(size_t)255;
        return p;
    };
    _Float16* xcl1   = (_Float16*)carve(XCL1_ELEMS * sizeof(_Float16));
    _Float16* w1r    = (_Float16*)carve(W1R_ELEMS  * sizeof(_Float16));
    _Float16* w2r    = (_Float16*)carve(W2R_ELEMS  * sizeof(_Float16));
    float*    pot1   = (float*)   carve(POT1_ELEMS * sizeof(float));
    float*    pooled = (float*)   carve(POOL1_ELEMS* sizeof(float));
    _Float16* xcl2   = (_Float16*)carve(XCL2_ELEMS * sizeof(_Float16));
    float*    pot2   = (float*)   carve(POT2_ELEMS * sizeof(float));

    const int B = 256;
    // Stage 0: repack inputs to f16 GEMM-friendly layouts
    pack_xcl1<<<(XCL1_ELEMS + B - 1) / B, B, 0, stream>>>(x, xcl1);
    pack_w1  <<<(W1R_ELEMS  + B - 1) / B, B, 0, stream>>>(w1, w1r);
    pack_w2  <<<(W2R_ELEMS  + B - 1) / B, B, 0, stream>>>(w2, w2r);

    // Stage 1: conv1 (WMMA) + fire(10)
    conv1_wmma<<<dim3(64, 2, T_STEPS), B, 0, stream>>>(xcl1, w1r, pot1);
    inhibit_inplace<<<1024, B, 0, stream>>>(pot1, 1024);

    // Stage 2: binarize + pool(2,2,1), inhibition on pooled patch (pad-47 region is identically zero)
    maxpool1<<<(T_STEPS * 256 * 289 + B - 1) / B, B, 0, stream>>>(pot1, pooled);
    inhibit_inplace<<<289, B, 0, stream>>>(pooled, 289);
    pack_xcl2<<<(XCL2_ELEMS + B - 1) / B, B, 0, stream>>>(pooled, xcl2);

    // Stage 3: conv2 (WMMA) on live 18x18 patch + fire(1), inhibition
    conv2_wmma<<<dim3(21, 2, T_STEPS), B, 0, stream>>>(xcl2, w2r, pot2);
    inhibit_inplace<<<324, B, 0, stream>>>(pot2, 324);

    // Stage 4: zero output, scatter 4x4/4 pooled + pad-2 live 6x6 region
    zero_f32<<<(T_STEPS * 256 * 1024 + B - 1) / B, B, 0, stream>>>(out, T_STEPS * 256 * 1024);
    final_pool<<<(T_STEPS * 256 * 36 + B - 1) / B, B, 0, stream>>>(pot2, out);
}
